// MoELayer_31275951849845
// MI455X (gfx1250) — compile-verified, matrix operand order checked by reading, not run on hardware
//
#include <hip/hip_runtime.h>
#include <hip/hip_bf16.h>

// MoE top-2 routed FFN for MI455X (gfx1250), wave32 + V_WMMA_F32_16X16X32_BF16.
// B=4 S=4096 D=256 E=8 H=512 K=2  ->  T=16384 tokens.

#define T_TOK 16384
#define D_DIM 256
#define E_NUM 8
#define H_DIM 512
#define MAX_TILES (T_TOK / 16)   // 1024 16-token tiles per expert worst case

typedef __bf16 v16bf __attribute__((ext_vector_type(16)));
typedef __bf16 v8bf  __attribute__((ext_vector_type(8)));
typedef float  v8f   __attribute__((ext_vector_type(8)));

// ---------------------------------------------------------------- zero init
__global__ __launch_bounds__(256) void moe_zero_kernel(float* __restrict__ out,
                                                       int* __restrict__ counts,
                                                       int n) {
  int i = blockIdx.x * 256 + threadIdx.x;
  if (i < n) out[i] = 0.0f;
  if (i < E_NUM) counts[i] = 0;
}

// ------------------------------------------------------------- conversions
__global__ __launch_bounds__(256) void moe_conv_x_kernel(const float* __restrict__ x,
                                                         __bf16* __restrict__ xb,
                                                         int n) {
  int i = blockIdx.x * 256 + threadIdx.x;
  if (i < n) xb[i] = (__bf16)x[i];
}

// W1 [E][D][H] (k-major) -> W1t [E][H][D]  (column n owns contiguous K run)
__global__ __launch_bounds__(256) void moe_conv_w1_kernel(const float* __restrict__ W1,
                                                          __bf16* __restrict__ W1t) {
  int i = blockIdx.x * 256 + threadIdx.x;             // i over E*H*D
  int e   = i / (D_DIM * H_DIM);
  int rem = i % (D_DIM * H_DIM);
  int n   = rem / D_DIM;                              // 0..H-1
  int k   = rem % D_DIM;                              // 0..D-1
  W1t[i] = (__bf16)W1[(size_t)e * D_DIM * H_DIM + (size_t)k * H_DIM + n];
}

// W2 [E][H][D] -> W2t [E][D][H]
__global__ __launch_bounds__(256) void moe_conv_w2_kernel(const float* __restrict__ W2,
                                                          __bf16* __restrict__ W2t) {
  int i = blockIdx.x * 256 + threadIdx.x;             // i over E*D*H
  int e   = i / (D_DIM * H_DIM);
  int rem = i % (D_DIM * H_DIM);
  int n   = rem / H_DIM;                              // 0..D-1
  int k   = rem % H_DIM;                              // 0..H-1
  W2t[i] = (__bf16)W2[(size_t)e * H_DIM * D_DIM + (size_t)k * D_DIM + n];
}

// ------------------------------------------------------------------- gate
__global__ __launch_bounds__(256) void moe_gate_kernel(const float* __restrict__ x,
                                                       const float* __restrict__ Wg,
                                                       const float* __restrict__ bg,
                                                       int* __restrict__ counts,
                                                       int* __restrict__ tokens) {
  __shared__ float wgs[D_DIM * E_NUM];                // 8 KB
  const int tid = threadIdx.x;
  for (int i = tid; i < D_DIM * E_NUM; i += 256) wgs[i] = Wg[i];
  __syncthreads();

  const int t = blockIdx.x * 256 + tid;
  float s[E_NUM];
#pragma unroll
  for (int e = 0; e < E_NUM; ++e) s[e] = bg[e];

  const float* xr = x + (size_t)t * D_DIM;
  for (int d = 0; d < D_DIM; ++d) {
    float xv = xr[d];
#pragma unroll
    for (int e = 0; e < E_NUM; ++e) s[e] += xv * wgs[d * E_NUM + e];
  }

  // top-2 (ties -> lowest index, matching jax.lax.top_k)
  int i1 = 0;
#pragma unroll
  for (int e = 1; e < E_NUM; ++e) if (s[e] > s[i1]) i1 = e;
  int i2 = -1;
#pragma unroll
  for (int e = 0; e < E_NUM; ++e)
    if (e != i1 && (i2 < 0 || s[e] > s[i2])) i2 = e;

  int p1 = atomicAdd(&counts[i1], 1);
  tokens[i1 * T_TOK + p1] = t;
  int p2 = atomicAdd(&counts[i2], 1);
  tokens[i2 * T_TOK + p2] = t;
}

// ------------------------------------------------------------ routed FFN
// One workgroup (8 wave32) per (expert, 16-token tile).
// GEMM1: [16xD] @ W1_e[DxH] -> gelu -> LDS   (each wave: 4 N-tiles of 16)
// GEMM2: [16xH] @ W2_e[HxD] -> atomic scatter (each wave: 2 N-tiles of 16)
__global__ __launch_bounds__(256) void moe_ffn_kernel(
    const __bf16* __restrict__ xb,     // [T][D]
    const __bf16* __restrict__ W1t,    // [E][H][D]
    const __bf16* __restrict__ W2t,    // [E][D][H]
    const float* __restrict__ b1,      // [E][H]
    const float* __restrict__ b2,      // [E][D]
    const int* __restrict__ counts,    // [E]
    const int* __restrict__ tokens,    // [E][T]
    float* __restrict__ out) {         // [T][D]
  __shared__ __bf16 xs[16 * D_DIM];    // 8 KB   activation tile
  __shared__ __bf16 hs[16 * H_DIM];    // 16 KB  hidden tile
  __shared__ int tokS[16];

  const int e    = blockIdx.x / MAX_TILES;
  const int tile = blockIdx.x % MAX_TILES;
  const int cnt  = counts[e];
  if (tile * 16 >= cnt) return;

  const int tid  = threadIdx.x;
  const int lane = tid & 31;
  const int wave = tid >> 5;
  const int lrow = lane & 15;          // row (A) / col (B,C) owned by lane
  const int lko  = (lane >> 4) * 8;    // K sub-offset AND accumulator M offset

  if (tid < 16) {
    int r = tile * 16 + tid;
    tokS[tid] = (r < cnt) ? tokens[e * T_TOK + r] : -1;
  }
  __syncthreads();

  // ---- gather 16 token rows of x (bf16) into LDS
  {
    int row = tid >> 4;                // 0..15
    int seg = tid & 15;                // 16 bf16 per thread
    __bf16* dst = &xs[row * D_DIM + seg * 16];
    int tk = tokS[row];
    if (tk >= 0) {
      const __bf16* src = xb + (size_t)tk * D_DIM + seg * 16;
      *(v8bf*)dst       = *(const v8bf*)src;
      *(v8bf*)(dst + 8) = *(const v8bf*)(src + 8);
    } else {
#pragma unroll
      for (int i = 0; i < 16; ++i) dst[i] = (__bf16)0.0f;
    }
  }
  __syncthreads();

  // ---- GEMM1: h = gelu(x @ W1_e + b1_e), K = D = 256
  v8f acc[4];
#pragma unroll
  for (int j = 0; j < 4; ++j)
#pragma unroll
    for (int i = 0; i < 8; ++i) acc[j][i] = 0.0f;

  const __bf16* w1e = W1t + (size_t)e * H_DIM * D_DIM;
  for (int kb = 0; kb < D_DIM; kb += 32) {
    v16bf afrag;
    {
      const __bf16* ap = &xs[lrow * D_DIM + kb + lko];
      v8bf lo = *(const v8bf*)ap;
      v8bf hi = *(const v8bf*)(ap + 16);
#pragma unroll
      for (int i = 0; i < 8; ++i) { afrag[i] = lo[i]; afrag[i + 8] = hi[i]; }
    }
#pragma unroll
    for (int j = 0; j < 4; ++j) {
      int ncol = (wave * 4 + j) * 16 + lrow;
      const __bf16* bp = w1e + (size_t)ncol * D_DIM + kb + lko;
      v16bf bfrag;
      v8bf lo = *(const v8bf*)bp;
      v8bf hi = *(const v8bf*)(bp + 16);
#pragma unroll
      for (int i = 0; i < 8; ++i) { bfrag[i] = lo[i]; bfrag[i + 8] = hi[i]; }
      acc[j] = __builtin_amdgcn_wmma_f32_16x16x32_bf16(
          false, afrag, false, bfrag, (short)0, acc[j], false, false);
    }
  }

  const float* b1e = b1 + e * H_DIM;
#pragma unroll
  for (int j = 0; j < 4; ++j) {
    int ncol = (wave * 4 + j) * 16 + lrow;
    float bias = b1e[ncol];
#pragma unroll
    for (int r = 0; r < 8; ++r) {
      int m = r + lko;                 // C-layout: VGPR r -> M = r + 8*(lane/16)
      float v = acc[j][r] + bias;
      float g = 0.5f * v * (1.0f + erff(v * 0.70710678118f));  // exact gelu
      hs[m * H_DIM + ncol] = (__bf16)g;
    }
  }
  __syncthreads();

  // ---- GEMM2: y = h @ W2_e + b2_e, K = H = 512
  v8f acc2[2];
#pragma unroll
  for (int j = 0; j < 2; ++j)
#pragma unroll
    for (int i = 0; i < 8; ++i) acc2[j][i] = 0.0f;

  const __bf16* w2e = W2t + (size_t)e * D_DIM * H_DIM;
  for (int kb = 0; kb < H_DIM; kb += 32) {
    v16bf afrag;
    {
      const __bf16* ap = &hs[lrow * H_DIM + kb + lko];
      v8bf lo = *(const v8bf*)ap;
      v8bf hi = *(const v8bf*)(ap + 16);
#pragma unroll
      for (int i = 0; i < 8; ++i) { afrag[i] = lo[i]; afrag[i + 8] = hi[i]; }
    }
#pragma unroll
    for (int j = 0; j < 2; ++j) {
      int ncol = (wave * 2 + j) * 16 + lrow;
      const __bf16* bp = w2e + (size_t)ncol * H_DIM + kb + lko;
      v16bf bfrag;
      v8bf lo = *(const v8bf*)bp;
      v8bf hi = *(const v8bf*)(bp + 16);
#pragma unroll
      for (int i = 0; i < 8; ++i) { bfrag[i] = lo[i]; bfrag[i + 8] = hi[i]; }
      acc2[j] = __builtin_amdgcn_wmma_f32_16x16x32_bf16(
          false, afrag, false, bfrag, (short)0, acc2[j], false, false);
    }
  }

  const float* b2e = b2 + e * D_DIM;
#pragma unroll
  for (int j = 0; j < 2; ++j) {
    int ncol = (wave * 2 + j) * 16 + lrow;
    float bias = b2e[ncol];
#pragma unroll
    for (int r = 0; r < 8; ++r) {
      int m = r + lko;
      int tk = tokS[m];
      if (tk >= 0)
        unsafeAtomicAdd(&out[(size_t)tk * D_DIM + ncol], acc2[j][r] + bias);
    }
  }
}

// ------------------------------------------------------------------ launch
extern "C" void kernel_launch(void* const* d_in, const int* in_sizes, int n_in,
                              void* d_out, int out_size, void* d_ws, size_t ws_size,
                              hipStream_t stream) {
  const float* x  = (const float*)d_in[0];   // [B,S,D]
  const float* Wg = (const float*)d_in[1];   // [D,E]
  const float* bg = (const float*)d_in[2];   // [E]
  const float* W1 = (const float*)d_in[3];   // [E,D,H]
  const float* b1 = (const float*)d_in[4];   // [E,H]
  const float* W2 = (const float*)d_in[5];   // [E,H,D]
  const float* b2 = (const float*)d_in[6];   // [E,D]
  float* out = (float*)d_out;                // [B,S,D] = [T,D]

  // workspace carving (~13.1 MB total)
  char* ws = (char*)d_ws;
  int*    counts = (int*)ws;                                   // 8 ints
  int*    tokens = (int*)(ws + 256);                           // E*T ints
  __bf16* xb     = (__bf16*)(ws + 256 + (size_t)E_NUM * T_TOK * sizeof(int));
  __bf16* W1t    = xb + (size_t)T_TOK * D_DIM;
  __bf16* W2t    = W1t + (size_t)E_NUM * D_DIM * H_DIM;

  const int nOut = T_TOK * D_DIM;            // 4 Mi
  const int nW   = E_NUM * D_DIM * H_DIM;    // 1 Mi

  moe_zero_kernel<<<(nOut + 255) / 256, 256, 0, stream>>>(out, counts, nOut);
  moe_conv_x_kernel<<<(nOut + 255) / 256, 256, 0, stream>>>(x, xb, nOut);
  moe_conv_w1_kernel<<<(nW + 255) / 256, 256, 0, stream>>>(W1, W1t);
  moe_conv_w2_kernel<<<(nW + 255) / 256, 256, 0, stream>>>(W2, W2t);
  moe_gate_kernel<<<T_TOK / 256, 256, 0, stream>>>(x, Wg, bg, counts, tokens);
  moe_ffn_kernel<<<E_NUM * MAX_TILES, 256, 0, stream>>>(xb, W1t, W2t, b1, b2,
                                                        counts, tokens, out);
}
